// StreamingRelPositionMultiHeadedAttention_31525059953359
// MI455X (gfx1250) — matrix operand order repaired and checked by reading, not compiled, not run on hardware
//
#include <hip/hip_runtime.h>
#include <math.h>

typedef __attribute__((ext_vector_type(16))) __bf16 v16bf;
typedef __attribute__((ext_vector_type(8)))  __bf16 v8bf;
typedef __attribute__((ext_vector_type(8)))  float  v8f;

#define BB   64
#define TT   128
#define FF   512
#define HH   8
#define DKK  64
#define LL   128
#define WW   384            // L + T + R
#define PPOS 511            // 2T-1+L+R
#define GTOT (LL + BB * TT) // 8320 rows in [cache ; kv_new]

__device__ __forceinline__ float4 ld4(const float* p) { return *(const float4*)p; }

__device__ __forceinline__ void pack8(v16bf& f, int o, float4 a, float4 b) {
    f[o + 0] = (__bf16)a.x; f[o + 1] = (__bf16)a.y;
    f[o + 2] = (__bf16)a.z; f[o + 3] = (__bf16)a.w;
    f[o + 4] = (__bf16)b.x; f[o + 5] = (__bf16)b.y;
    f[o + 6] = (__bf16)b.z; f[o + 7] = (__bf16)b.w;
}
__device__ __forceinline__ void pack8add(v16bf& f, int o, float4 a, float4 b,
                                         float4 c, float4 d) {
    f[o + 0] = (__bf16)(a.x + c.x); f[o + 1] = (__bf16)(a.y + c.y);
    f[o + 2] = (__bf16)(a.z + c.z); f[o + 3] = (__bf16)(a.w + c.w);
    f[o + 4] = (__bf16)(b.x + d.x); f[o + 5] = (__bf16)(b.y + d.y);
    f[o + 6] = (__bf16)(b.z + d.z); f[o + 7] = (__bf16)(b.w + d.w);
}

// ---------------------------------------------------------------------------
// GEMM: C[M,N] = A[M,K] @ B[K,N] + bias.
// 128 threads = 4 waves; block tile 64x64. Each wave: one A fragment feeds
// FOUR 16x16 C tiles (A reuse x4); staged 32x64 bf16 B chunk in LDS feeds
// 16 WMMAs per barrier pair. Out-of-range A rows CLAMPED; stores guarded.
// ---------------------------------------------------------------------------
__global__ __launch_bounds__(128) void gemm_bf16_wmma(
    const float* __restrict__ A, const float* __restrict__ Bm,
    const float* __restrict__ bias, float* __restrict__ C,
    int M, int Kdim, int N) {
    __shared__ __bf16 BsT[64][32];   // [col][krow] transposed chunk (4 KB)

    const int tiles_n = N >> 6;                 // 64-col panels
    const int tn  = blockIdx.x % tiles_n;
    const int tm0 = (blockIdx.x / tiles_n) * 64;
    const int nc0 = tn * 64;
    const int tid = threadIdx.x;
    const int wv = tid >> 5, lane = tid & 31;
    const int l16 = lane & 15, kh = lane >> 4;
    const int row  = tm0 + wv * 16 + l16;
    const int rowc = min(row, M - 1);
    const float* __restrict__ Arow = A + (size_t)rowc * Kdim;

    // staging decomposition: 2 K-rows x 8 cols per thread (b32-packed stores)
    const int sr0 = (tid & 15) * 2;   // K row pair 0..30
    const int sc0 = (tid >> 4) * 8;   // col group 0..56

    v8f acc[4] = {};
    for (int k0 = 0; k0 < Kdim; k0 += 32) {
        {
            const float* b0 = &Bm[(size_t)(k0 + sr0) * N + nc0 + sc0];
            const float* b1 = b0 + N;
            float4 x0 = ld4(b0), x1 = ld4(b0 + 4);
            float4 y0 = ld4(b1), y1 = ld4(b1 + 4);
            float xs[8] = {x0.x, x0.y, x0.z, x0.w, x1.x, x1.y, x1.z, x1.w};
            float ys[8] = {y0.x, y0.y, y0.z, y0.w, y1.x, y1.y, y1.z, y1.w};
#pragma unroll
            for (int c = 0; c < 8; ++c) {
                union { __bf16 h[2]; unsigned int u; } pk;
                pk.h[0] = (__bf16)xs[c];
                pk.h[1] = (__bf16)ys[c];
                *(unsigned int*)&BsT[sc0 + c][sr0] = pk.u;
            }
        }
        // A-fragment: one contiguous 16-float run per lane
        float4 a0 = ld4(&Arow[k0 + 16 * kh]);
        float4 a1 = ld4(&Arow[k0 + 16 * kh + 4]);
        float4 a2 = ld4(&Arow[k0 + 16 * kh + 8]);
        float4 a3 = ld4(&Arow[k0 + 16 * kh + 12]);
        if (k0 + 32 < Kdim) __builtin_prefetch(&Arow[k0 + 32], 0, 1);
        __syncthreads();
        v16bf af;
        pack8(af, 0, a0, a1);
        pack8(af, 8, a2, a3);
#pragma unroll
        for (int ct = 0; ct < 4; ++ct) {
            v16bf bf_ = *(const v16bf*)&BsT[ct * 16 + l16][16 * kh];
            acc[ct] = __builtin_amdgcn_wmma_f32_16x16x32_bf16(
                false, af, false, bf_, (short)0, acc[ct], false, false);
        }
        __syncthreads();
    }
#pragma unroll
    for (int ct = 0; ct < 4; ++ct) {
        const int col = nc0 + ct * 16 + l16;
        const float bv = bias ? bias[col] : 0.0f;
#pragma unroll
        for (int r = 0; r < 8; ++r) {
            int rr = tm0 + wv * 16 + r + 8 * kh;
            if (rr < M) C[(size_t)rr * N + col] = acc[ct][r] + bv;
        }
    }
}

// ---------------------------------------------------------------------------
// Attention: one wave32 per (b, h, 16-row t-tile).
// ---------------------------------------------------------------------------
__global__ __launch_bounds__(32) void attn_wmma(
    const float* __restrict__ q, const float* __restrict__ kmat,
    const float* __restrict__ vmat, const float* __restrict__ cache,
    const float* __restrict__ pm, const unsigned char* __restrict__ mask,
    const float* __restrict__ bias_u, const float* __restrict__ bias_v,
    float* __restrict__ xout) {
    __shared__ float  G[16][512];     // rel-pos term q_v @ p^T
    __shared__ __bf16 Ks[32][64];     // staged k window rows (bf16)
    __shared__ __bf16 Vs[32][64];     // staged v window rows (bf16)
    __shared__ __bf16 Ps[16][32];     // probability tile relayout

    const int blk  = blockIdx.x;
    const int t0   = (blk & 7) * 16;      // T/16 = 8 tiles
    const int h    = (blk >> 3) & 7;
    const int b    = blk >> 6;
    const int lane = threadIdx.x & 31;
    const int l16  = lane & 15;
    const int kh   = lane >> 4;

    // --- q_u / q_v A-fragments (contiguous b128 loads) ---
    const float* __restrict__ qp = q + (size_t)(b * TT + t0 + l16) * FF + h * DKK;
    const float* __restrict__ up = bias_u + h * DKK;
    const float* __restrict__ vp = bias_v + h * DKK;
    v16bf qu[2], qv[2];
#pragma unroll
    for (int ks = 0; ks < 2; ++ks) {
        int d0 = ks * 32 + 16 * kh;
        int d1 = ks * 32 + 16 * kh + 8;
        float4 qa = ld4(qp + d0), qb = ld4(qp + d0 + 4);
        float4 qc = ld4(qp + d1), qd = ld4(qp + d1 + 4);
        float4 ua = ld4(up + d0), ub = ld4(up + d0 + 4);
        float4 uc = ld4(up + d1), ud = ld4(up + d1 + 4);
        float4 va = ld4(vp + d0), vb = ld4(vp + d0 + 4);
        float4 vc = ld4(vp + d1), vd = ld4(vp + d1 + 4);
        pack8add(qu[ks], 0, qa, qb, ua, ub);
        pack8add(qu[ks], 8, qc, qd, uc, ud);
        pack8add(qv[ks], 0, qa, qb, va, vb);
        pack8add(qv[ks], 8, qc, qd, vc, vd);
    }

    // --- Phase 1: G = q_v @ p^T (column 511 clamped; never read) ---
    for (int pt = 0; pt < 32; ++pt) {
        const int ppc = min(pt * 16 + l16, PPOS - 1);   // branch-free clamp
        const float* __restrict__ pr = pm + (size_t)ppc * FF + h * DKK;
        v8f acc = {};
#pragma unroll
        for (int ks = 0; ks < 2; ++ks) {
            int d0 = ks * 32 + 16 * kh;
            float4 p0 = ld4(pr + d0),     p1 = ld4(pr + d0 + 4);
            float4 p2 = ld4(pr + d0 + 8), p3 = ld4(pr + d0 + 12);
            v16bf bf_;
            pack8(bf_, 0, p0, p1);
            pack8(bf_, 8, p2, p3);
            acc = __builtin_amdgcn_wmma_f32_16x16x32_bf16(
                false, qv[ks], false, bf_, (short)0, acc, false, false);
        }
#pragma unroll
        for (int r = 0; r < 8; ++r) G[r + 8 * kh][pt * 16 + l16] = acc[r];
    }

    // --- Phase 2: flash attention over the 384-wide window ---
    float mrun[8], lrun[8];
#pragma unroll
    for (int r = 0; r < 8; ++r) { mrun[r] = -1e30f; lrun[r] = 0.0f; }
    v8f O[4] = {};
    const int shiftbase = TT - 1 - t0;   // p index = shiftbase - m + w

    for (int w0 = 0; w0 < WW; w0 += 32) {
        // stage 32 k/v window rows: one row per lane, contiguous b128 loads,
        // region select is branch-free (boundaries are 16-aligned)
        {
            const int g   = b * TT + w0 + lane;
            const bool inC = g < LL;
            const int  gc  = min(g, GTOT - 1);
            const float zm = (g < GTOT) ? 1.0f : 0.0f;
            const float* __restrict__ krow =
                inC ? (cache + (size_t)g * (HH * 2 * DKK) + h * (2 * DKK))
                    : (kmat + (size_t)(gc - LL) * FF + h * DKK);
            const float* __restrict__ vrow =
                inC ? (cache + (size_t)g * (HH * 2 * DKK) + h * (2 * DKK) + DKK)
                    : (vmat + (size_t)(gc - LL) * FF + h * DKK);
#pragma unroll
            for (int qd = 0; qd < 16; ++qd) {
                float4 kf = ld4(krow + 4 * qd);
                float4 vf = ld4(vrow + 4 * qd);
                Ks[lane][4 * qd + 0] = (__bf16)(kf.x * zm);
                Ks[lane][4 * qd + 1] = (__bf16)(kf.y * zm);
                Ks[lane][4 * qd + 2] = (__bf16)(kf.z * zm);
                Ks[lane][4 * qd + 3] = (__bf16)(kf.w * zm);
                Vs[lane][4 * qd + 0] = (__bf16)(vf.x * zm);
                Vs[lane][4 * qd + 1] = (__bf16)(vf.y * zm);
                Vs[lane][4 * qd + 2] = (__bf16)(vf.z * zm);
                Vs[lane][4 * qd + 3] = (__bf16)(vf.w * zm);
            }
        }
        // AC scores for two 16-col subtiles (same-wave LDS is in-order)
        float s[2][8];
#pragma unroll
        for (int st = 0; st < 2; ++st) {
            v8f acc = {};
#pragma unroll
            for (int ks = 0; ks < 2; ++ks) {
                v16bf bf_ = *(const v16bf*)&Ks[st * 16 + l16][ks * 32 + 16 * kh];
                acc = __builtin_amdgcn_wmma_f32_16x16x32_bf16(
                    false, qu[ks], false, bf_, (short)0, acc, false, false);
            }
            const int wcol = w0 + st * 16 + l16;
            const bool mok = mask[b * WW + wcol] != 0;
#pragma unroll
            for (int r = 0; r < 8; ++r) {
                int m = r + 8 * kh;
                float g = G[m][shiftbase - m + wcol];
                s[st][r] = mok ? (acc[r] + g) * 0.125f : -1e30f;
            }
        }
        // online softmax across the 32 columns
#pragma unroll
        for (int r = 0; r < 8; ++r) {
            float rm = fmaxf(s[0][r], s[1][r]);
#pragma unroll
            for (int xm = 8; xm >= 1; xm >>= 1)
                rm = fmaxf(rm, __shfl_xor(rm, xm, 32));
            float mnew = fmaxf(mrun[r], rm);
            float corr = __expf(mrun[r] - mnew);
            float p0 = (s[0][r] < -1e29f) ? 0.0f : __expf(s[0][r] - mnew);
            float p1 = (s[1][r] < -1e29f) ? 0.0f : __expf(s[1][r] - mnew);
            float rs = p0 + p1;
#pragma unroll
            for (int xm = 8; xm >= 1; xm >>= 1)
                rs += __shfl_xor(rs, xm, 32);
            lrun[r] = lrun[r] * corr + rs;
            mrun[r] = mnew;
#pragma unroll
            for (int dt = 0; dt < 4; ++dt) O[dt][r] *= corr;
            Ps[r + 8 * kh][l16]      = (__bf16)p0;
            Ps[r + 8 * kh][16 + l16] = (__bf16)p1;
        }
        // re-layout P as a 16x32 A-fragment (two contiguous 16B LDS reads)
        v16bf pa;
        {
            v8bf lo = *(const v8bf*)&Ps[l16][16 * kh];
            v8bf hi = *(const v8bf*)&Ps[l16][16 * kh + 8];
#pragma unroll
            for (int e = 0; e < 8; ++e) { pa[e] = lo[e]; pa[8 + e] = hi[e]; }
        }
        // O += P @ v_w over 4 d-tiles
#pragma unroll
        for (int dt = 0; dt < 4; ++dt) {
            v16bf vb;
#pragma unroll
            for (int e = 0; e < 16; ++e) vb[e] = Vs[16 * kh + e][dt * 16 + l16];
            O[dt] = __builtin_amdgcn_wmma_f32_16x16x32_bf16(
                false, pa, false, vb, (short)0, O[dt], false, false);
        }
    }

    // --- normalize and write x (b, t, h*dk) ---
#pragma unroll
    for (int r = 0; r < 8; ++r) {
        float inv = (lrun[r] > 0.0f) ? (1.0f / lrun[r]) : 0.0f;
        int trow = t0 + r + 8 * kh;
#pragma unroll
        for (int dt = 0; dt < 4; ++dt)
            xout[(size_t)(b * TT + trow) * FF + h * DKK + dt * 16 + l16] =
                O[dt][r] * inv;
    }
}

// new_cache[i][h][c] = {k|v}[b=63, t=i, h, c]
__global__ void cache_copy(const float* __restrict__ k,
                           const float* __restrict__ v,
                           float* __restrict__ nc) {
    int i = blockIdx.x * blockDim.x + threadIdx.x;   // float4 index
    if (i >= LL * HH * 2 * DKK / 4) return;
    int c4  = (i & 31) * 4;
    int h   = (i >> 5) & (HH - 1);
    int row = i >> 8;
    const float* src = ((c4 < DKK) ? k : v) +
                       (size_t)((BB - 1) * TT + row) * FF + h * DKK +
                       (c4 & (DKK - 1));
    ((float4*)nc)[i] = ld4(src);
}

extern "C" void kernel_launch(void* const* d_in, const int* in_sizes, int n_in,
                              void* d_out, int out_size, void* d_ws, size_t ws_size,
                              hipStream_t stream) {
    (void)in_sizes; (void)n_in; (void)out_size; (void)ws_size;
    const float* query   = (const float*)d_in[0];
    const float* keyi    = (const float*)d_in[1];
    const float* value   = (const float*)d_in[2];
    const unsigned char* mask = (const unsigned char*)d_in[3];
    const float* pos_emb = (const float*)d_in[4];
    const float* cache   = (const float*)d_in[5];
    const float* W_q  = (const float*)d_in[6];
    const float* b_q  = (const float*)d_in[7];
    const float* W_k  = (const float*)d_in[8];
    const float* b_k  = (const float*)d_in[9];
    const float* W_v  = (const float*)d_in[10];
    const float* b_v  = (const float*)d_in[11];
    const float* W_pos = (const float*)d_in[12];
    const float* bias_u = (const float*)d_in[13];
    const float* bias_v = (const float*)d_in[14];
    const float* W_out = (const float*)d_in[15];
    const float* b_out = (const float*)d_in[16];

    const int MQ = BB * TT;                 // 8192
    float* ws  = (float*)d_ws;
    float* qw  = ws;
    float* kw  = qw + (size_t)MQ * FF;
    float* vw  = kw + (size_t)MQ * FF;
    float* pmw = vw + (size_t)MQ * FF;      // 512x512 (511 used)
    float* xw  = pmw + (size_t)512 * FF;

    float* outp = (float*)d_out;            // (B,T,F)
    float* ncp  = outp + (size_t)MQ * FF;   // new_cache (L,H,2DK)

    dim3 blk4(128), wave(32);
    const int blocksQ = (MQ / 64) * (FF / 64);          // 128*8
    gemm_bf16_wmma<<<dim3(blocksQ), blk4, 0, stream>>>(query, W_q, b_q, qw, MQ, FF, FF);
    gemm_bf16_wmma<<<dim3(blocksQ), blk4, 0, stream>>>(keyi,  W_k, b_k, kw, MQ, FF, FF);
    gemm_bf16_wmma<<<dim3(blocksQ), blk4, 0, stream>>>(value, W_v, b_v, vw, MQ, FF, FF);

    const int blocksP = ((PPOS + 63) / 64) * (FF / 64); // 8*8
    gemm_bf16_wmma<<<dim3(blocksP), blk4, 0, stream>>>(pos_emb, W_pos, nullptr, pmw,
                                                       PPOS, FF, FF);

    attn_wmma<<<dim3(BB * HH * (TT / 16)), wave, 0, stream>>>(
        qw, kw, vw, cache, pmw, mask, bias_u, bias_v, xw);

    gemm_bf16_wmma<<<dim3(blocksQ), blk4, 0, stream>>>(xw, W_out, b_out, outp, MQ, FF, FF);

    cache_copy<<<dim3((LL * HH * 2 * DKK / 4 + 255) / 256), dim3(256), 0, stream>>>(
        kw, vw, ncp);
}
